// TransformerEncoderLayer_22273700397707
// MI455X (gfx1250) — matrix operand order repaired and checked by reading, not compiled, not run on hardware
//
#include <hip/hip_runtime.h>
#include <hip/hip_bf16.h>

// ---------------------------------------------------------------------------
// TransformerEncoderLayer for MI455X (gfx1250, wave32).
//   0) one-time weight transpose+f16 (B tiles become async-stageable)
//   1) LN1(x) -> h (f16)
//   2) h @ Wqkv + bqkv -> qkv (f16)          [WMMA, fully-async double buffer]
//   3) flash attention (online softmax)      [WMMA, 4 waves share K/V/mask LDS]
//   4) o @ Wo + bo + x -> x1 (f32)           [WMMA, fused residual]
//   5) lnf(ln2(x1)) -> h2 (f16)
//   6) gelu(h2 @ W1 + b1) -> g (f16)         [WMMA, fused exact GELU]
//   7) g @ W2 + b2 + x1 -> out (f32)         [WMMA, fused residual]
// ---------------------------------------------------------------------------

typedef _Float16 half_t;
typedef __attribute__((ext_vector_type(2)))  _Float16 v2h;
typedef __attribute__((ext_vector_type(8)))  float    v8f;
typedef __attribute__((ext_vector_type(8)))  _Float16 v8h;
typedef __attribute__((ext_vector_type(16))) _Float16 v16h;

#define SEQ   2048
#define EMB   1024
#define HEADS 16
#define HDIM  64
#define DFF_  4096
#define ROWS  (2 * SEQ)          // N*L = 4096

static __device__ __forceinline__ v8f wmma16(v16h a, v16h b, v8f c) {
  return __builtin_amdgcn_wmma_f32_16x16x32_f16(false, a, false, b, (short)0, c,
                                                false, false);
}

static __device__ __forceinline__ v16h cat8(v8h lo, v8h hi) {
  return __builtin_shufflevector(lo, hi, 0, 1, 2, 3, 4, 5, 6, 7, 8, 9, 10, 11,
                                 12, 13, 14, 15);
}

// CDNA5 async global->LDS copy (16B per lane), tracked by ASYNCcnt.
static __device__ __forceinline__ void async_cp16(void* lds, const void* g) {
  unsigned l = (unsigned)(size_t)lds;           // low 32 bits = LDS offset
  unsigned long long a = (unsigned long long)(size_t)g;
  asm volatile("global_load_async_to_lds_b128 %0, %1, off"
               :: "v"(l), "v"(a) : "memory");
}
static __device__ __forceinline__ void async_wait0() {
  asm volatile("s_wait_asynccnt 0x0" ::: "memory");
}
// Wave-local LDS ordering (per-wave DS ops are in-order; fence the compiler
// and drain DScnt so own-wave cross-lane LDS data is ready).
static __device__ __forceinline__ void lds_fence_wave() {
  asm volatile("s_wait_dscnt 0x0" ::: "memory");
}

static __device__ __forceinline__ float block_sum256(float val, float* red) {
  const int tid = threadIdx.x;
  red[tid] = val;
  __syncthreads();
  for (int st = 128; st > 0; st >>= 1) {
    if (tid < st) red[tid] += red[tid + st];
    __syncthreads();
  }
  float r = red[0];
  __syncthreads();
  return r;
}

// ---------------------------------------------------------------------------
// Weight transpose + f32->f16: out[n][k] = (f16) in[k][n]
// ---------------------------------------------------------------------------
__global__ __launch_bounds__(256) void transpose_f32_to_f16_kernel(
    const float* __restrict__ in, half_t* __restrict__ out, int K, int N) {
  __shared__ float tile[32][33];
  const int kb = blockIdx.y * 32, nb = blockIdx.x * 32;
  const int tx = threadIdx.x & 31, ty = threadIdx.x >> 5;  // 32x8
#pragma unroll
  for (int j = 0; j < 32; j += 8)
    tile[ty + j][tx] = in[(size_t)(kb + ty + j) * N + nb + tx];
  __syncthreads();
#pragma unroll
  for (int j = 0; j < 32; j += 8)
    out[(size_t)(nb + ty + j) * K + kb + tx] = (half_t)tile[tx][ty + j];
}

// ---------------------------------------------------------------------------
// LayerNorm over E=1024, one block (256 threads) per row, f16 output
// ---------------------------------------------------------------------------
__global__ __launch_bounds__(256) void layernorm_kernel(
    const float* __restrict__ x, const float* __restrict__ s,
    const float* __restrict__ b, half_t* __restrict__ out) {
  __shared__ float red[256];
  const int tid = threadIdx.x;
  const float* xr = x + (size_t)blockIdx.x * EMB;
  half_t* orow = out + (size_t)blockIdx.x * EMB;
  float v[4];
  float sum = 0.f;
#pragma unroll
  for (int j = 0; j < 4; ++j) { v[j] = xr[tid + j * 256]; sum += v[j]; }
  float mu = block_sum256(sum, red) * (1.0f / EMB);
  float sq = 0.f;
#pragma unroll
  for (int j = 0; j < 4; ++j) { float d = v[j] - mu; sq += d * d; }
  float rstd = rsqrtf(block_sum256(sq, red) * (1.0f / EMB) + 1e-5f);
#pragma unroll
  for (int j = 0; j < 4; ++j) {
    int idx = tid + j * 256;
    orow[idx] = (half_t)((v[j] - mu) * rstd * s[idx] + b[idx]);
  }
}

// lnf(ln2(x)) fused, f16 output
__global__ __launch_bounds__(256) void double_ln_kernel(
    const float* __restrict__ x, const float* __restrict__ s2,
    const float* __restrict__ b2, const float* __restrict__ sf,
    const float* __restrict__ bf, half_t* __restrict__ out) {
  __shared__ float red[256];
  const int tid = threadIdx.x;
  const float* xr = x + (size_t)blockIdx.x * EMB;
  half_t* orow = out + (size_t)blockIdx.x * EMB;
  float v[4];
  float sum = 0.f;
#pragma unroll
  for (int j = 0; j < 4; ++j) { v[j] = xr[tid + j * 256]; sum += v[j]; }
  float mu = block_sum256(sum, red) * (1.0f / EMB);
  float sq = 0.f;
#pragma unroll
  for (int j = 0; j < 4; ++j) { float d = v[j] - mu; sq += d * d; }
  float rstd = rsqrtf(block_sum256(sq, red) * (1.0f / EMB) + 1e-5f);
  float y[4];
  float sum2 = 0.f;
#pragma unroll
  for (int j = 0; j < 4; ++j) {
    int idx = tid + j * 256;
    y[j] = (v[j] - mu) * rstd * s2[idx] + b2[idx];
    sum2 += y[j];
  }
  float mu2 = block_sum256(sum2, red) * (1.0f / EMB);
  float sq2 = 0.f;
#pragma unroll
  for (int j = 0; j < 4; ++j) { float d = y[j] - mu2; sq2 += d * d; }
  float rstd2 = rsqrtf(block_sum256(sq2, red) * (1.0f / EMB) + 1e-5f);
#pragma unroll
  for (int j = 0; j < 4; ++j) {
    int idx = tid + j * 256;
    orow[idx] = (half_t)((y[j] - mu2) * rstd2 * sf[idx] + bf[idx]);
  }
}

// ---------------------------------------------------------------------------
// WMMA f16 GEMM: C[M,N] = A[M,K] @ Bt[N,K]^T + bias, fused epilogue.
// 256 threads = 8 wave32s; block tile 128x128; wave tile 32x64; K-step 64.
// Both operands staged with async global->LDS; double-buffered; one barrier
// per 16 WMMAs per wave.
// ---------------------------------------------------------------------------
enum { EPI_F16 = 0, EPI_GELU_F16 = 1, EPI_RES_F32 = 2 };

template <int EPI>
__global__ __launch_bounds__(256) void gemm_f16_kernel(
    const half_t* __restrict__ A, const half_t* __restrict__ Bt,
    const float* __restrict__ bias, const float* __restrict__ res,
    void* __restrict__ out, int M, int K, int Ncols) {
  __shared__ half_t As[2][128][72];   // row-major, K contiguous
  __shared__ half_t Bs[2][128][72];   // N-major,  K contiguous
  const int tid  = threadIdx.x;
  const int lane = tid & 31;
  const int wid  = tid >> 5;
  const int wm   = wid >> 1;          // 0..3 -> 32-row slice
  const int wn   = wid & 1;           // 0..1 -> 64-col slice
  const int bm   = blockIdx.y * 128;
  const int bn   = blockIdx.x * 128;
  const int lrow = lane & 15;
  const int lhi  = lane >> 4;

  v8f acc[8] = {};

  const int srow = tid >> 1;          // 0..127
  const int sc   = (tid & 1) * 32;    // 0 or 32
  auto stage = [&](int buf, int kb) {
    const half_t* sa = A + (size_t)(bm + srow) * K + kb + sc;
    const half_t* sb = Bt + (size_t)(bn + srow) * K + kb + sc;
#pragma unroll
    for (int j = 0; j < 4; ++j) {
      async_cp16(&As[buf][srow][sc + j * 8], sa + j * 8);
      async_cp16(&Bs[buf][srow][sc + j * 8], sb + j * 8);
    }
  };

  stage(0, 0);

  int buf = 0;
  for (int kb = 0; kb < K; kb += 64, buf ^= 1) {
    async_wait0();
    __syncthreads();
    if (kb + 64 < K) stage(buf ^ 1, kb + 64);

    const int aoff = lhi * 8;
    const int boff = lhi * 16;
    const int ar0  = wm * 32 + lrow;
#pragma unroll
    for (int kc = 0; kc < 2; ++kc) {
      const int ko = kc * 32;
      v16h a0 = cat8(*(const v8h*)&As[buf][ar0][ko + aoff],
                     *(const v8h*)&As[buf][ar0][ko + 16 + aoff]);
      v16h a1 = cat8(*(const v8h*)&As[buf][ar0 + 16][ko + aoff],
                     *(const v8h*)&As[buf][ar0 + 16][ko + 16 + aoff]);
#pragma unroll
      for (int nf = 0; nf < 4; ++nf) {
        const int bcol = wn * 64 + nf * 16 + lrow;
        v16h bfrag = cat8(*(const v8h*)&Bs[buf][bcol][ko + aoff],
                          *(const v8h*)&Bs[buf][bcol][ko + 16 + aoff]);
        acc[nf]     = wmma16(a0, bfrag, acc[nf]);
        acc[4 + nf] = wmma16(a1, bfrag, acc[4 + nf]);
      }
    }
  }

  // Epilogue: C layout — VGPR r holds row r (lanes 0-15) / row r+8 (lanes 16-31)
#pragma unroll
  for (int nf = 0; nf < 4; ++nf) {
    const int col = bn + wn * 64 + nf * 16 + lrow;
    const float bv = bias[col];
#pragma unroll
    for (int hf = 0; hf < 2; ++hf) {
#pragma unroll
      for (int r = 0; r < 8; ++r) {
        const int row = bm + wm * 32 + hf * 16 + r + lhi * 8;
        float v = acc[hf * 4 + nf][r] + bv;
        const size_t idx = (size_t)row * Ncols + col;
        if (EPI == EPI_GELU_F16) {
          float g = 0.5f * v * (1.0f + erff(v * 0.70710678118654752f));
          ((half_t*)out)[idx] = (half_t)g;
        } else if (EPI == EPI_RES_F32) {
          ((float*)out)[idx] = v + res[idx];
        } else {
          ((half_t*)out)[idx] = (half_t)v;
        }
      }
    }
  }
}

// ---------------------------------------------------------------------------
// Flash attention: 128 threads = 4 waves; block handles 128 q-rows of one
// (n, h); waves share K/V/mask LDS stage; each wave owns 32 q-rows processed
// as two 16-row groups. S column-tiles are interleaved (tile b covers
// k = 2c+b) so the two probabilities per row pack into one b32 LDS store.
// ---------------------------------------------------------------------------
__global__ __launch_bounds__(128) void flash_attn_kernel(
    const half_t* __restrict__ qkv,          // [N, L, 3E] f16
    const unsigned char* __restrict__ mask,  // [N, H, L, L] bool
    half_t* __restrict__ o) {                // [N, L, E] f16
  const int qb   = blockIdx.x * 128;
  const int h    = blockIdx.y;
  const int n    = blockIdx.z;
  const int tid  = threadIdx.x;
  const int lane = tid & 31;
  const int w    = tid >> 5;     // wave id 0..3
  const int lrow = lane & 15;
  const int lhi  = lane >> 4;

  __shared__ half_t Qs[128][72];
  __shared__ half_t Ks[32][72];
  __shared__ half_t Vt[64][40];            // transposed V: [d][k]
  __shared__ unsigned char Mst[32][136];   // transposed mask: [kcol][qrow]
  __shared__ half_t Ps[4][16][40];         // per-wave P, plain [row][k]

  const size_t rstride = 3 * EMB;
  const half_t* qsrc = qkv + ((size_t)n * SEQ + qb) * rstride + h * HDIM;

  // Stage Q tile 128x64 (async, 8x16B per thread: one row per thread)
  {
    const half_t* src = qsrc + (size_t)tid * rstride;
#pragma unroll
    for (int j = 0; j < 8; ++j)
      async_cp16(&Qs[tid][j * 8], src + j * 8);
  }
  async_wait0();
  __syncthreads();

  // Per-wave Q fragments: 2 row-groups x 2 d-chunks
  v16h aq[2][2];
#pragma unroll
  for (int g = 0; g < 2; ++g) {
    const int qr = w * 32 + g * 16 + lrow;
#pragma unroll
    for (int c = 0; c < 2; ++c)
      aq[g][c] = cat8(*(const v8h*)&Qs[qr][c * 32 + lhi * 8],
                      *(const v8h*)&Qs[qr][c * 32 + 16 + lhi * 8]);
  }

  float m[2][8], l[2][8];
  v8f Oacc[2][4] = {};
#pragma unroll
  for (int g = 0; g < 2; ++g)
#pragma unroll
    for (int r = 0; r < 8; ++r) { m[g][r] = -1e30f; l[g][r] = 0.f; }

  const unsigned char* mblock =
      mask + (((size_t)n * HEADS + h) * SEQ + qb) * SEQ;

  for (int kb = 0; kb < SEQ; kb += 32) {
    // Stage K rows [kb,kb+32) row-major (async, 2x16B per thread)
    {
      int row = tid >> 2, c = (tid & 3) * 16;
      const half_t* src =
          qkv + ((size_t)n * SEQ + kb + row) * rstride + EMB + h * HDIM + c;
      async_cp16(&Ks[row][c], src);
      async_cp16(&Ks[row][c + 8], src + 8);
    }
    // Stage mask tile transposed: Mst[kcol][qrow] (coalesced 32B global read)
    {
      const unsigned long long* ms =
          (const unsigned long long*)(mblock + (size_t)tid * SEQ + kb);
#pragma unroll
      for (int q = 0; q < 4; ++q) {
        unsigned long long mv = ms[q];
#pragma unroll
        for (int j = 0; j < 8; ++j)
          Mst[q * 8 + j][tid] = (unsigned char)(mv >> (8 * j));
      }
    }
    // Stage V transposed into Vt[d][k] (regular loads + scatter ds stores)
    {
      int row = tid >> 2, dch = (tid & 3) * 16;
      const half_t* src =
          qkv + ((size_t)n * SEQ + kb + row) * rstride + 2 * EMB + h * HDIM + dch;
#pragma unroll
      for (int jj = 0; jj < 2; ++jj) {
        v8h vv = *(const v8h*)(src + jj * 8);
#pragma unroll
        for (int e = 0; e < 8; ++e) Vt[dch + jj * 8 + e][row] = vv[e];
      }
    }
    async_wait0();
    __syncthreads();

#pragma unroll
    for (int g = 0; g < 2; ++g) {
      // S = Q K^T; tile b covers k-columns {2c+b}; contraction over d chained
      v8f s[2];
#pragma unroll
      for (int cb = 0; cb < 2; ++cb) {
        const int krow = 2 * lrow + cb;
        v16h b0 = cat8(*(const v8h*)&Ks[krow][lhi * 8],
                       *(const v8h*)&Ks[krow][16 + lhi * 8]);
        v16h b1 = cat8(*(const v8h*)&Ks[krow][32 + lhi * 8],
                       *(const v8h*)&Ks[krow][32 + 16 + lhi * 8]);
        v8f t = {};
        t = wmma16(aq[g][0], b0, t);
        t = wmma16(aq[g][1], b1, t);
        s[cb] = t;
      }

      // Mask (one b64 per tile per lane), scale, online-softmax statistics
      const int qoff = w * 32 + g * 16 + lhi * 8;
      const unsigned long long mq0 =
          *(const unsigned long long*)&Mst[2 * lrow + 0][qoff];
      const unsigned long long mq1 =
          *(const unsigned long long*)&Mst[2 * lrow + 1][qoff];
      float mnew[8];
#pragma unroll
      for (int r = 0; r < 8; ++r) {
        float s0 = s[0][r] * 0.125f;
        float s1 = s[1][r] * 0.125f;
        if ((mq0 >> (8 * r)) & 0xff) s0 = -1e30f;
        if ((mq1 >> (8 * r)) & 0xff) s1 = -1e30f;
        s[0][r] = s0;
        s[1][r] = s1;
        float rm = fmaxf(s0, s1);
        rm = fmaxf(rm, __shfl_xor(rm, 1, 32));
        rm = fmaxf(rm, __shfl_xor(rm, 2, 32));
        rm = fmaxf(rm, __shfl_xor(rm, 4, 32));
        rm = fmaxf(rm, __shfl_xor(rm, 8, 32));
        mnew[r] = fmaxf(m[g][r], rm);
      }
#pragma unroll
      for (int r = 0; r < 8; ++r) {
        const float alpha = __expf(m[g][r] - mnew[r]);
        const float p0 = __expf(s[0][r] - mnew[r]);
        const float p1 = __expf(s[1][r] - mnew[r]);
        const int prow = r + lhi * 8;
        v2h pp;
        pp[0] = (half_t)p0;
        pp[1] = (half_t)p1;
        *(v2h*)&Ps[w][prow][2 * lrow] = pp;   // k = 2*lrow(+1): plain layout
        float rs = p0 + p1;
        rs += __shfl_xor(rs, 1, 32);
        rs += __shfl_xor(rs, 2, 32);
        rs += __shfl_xor(rs, 4, 32);
        rs += __shfl_xor(rs, 8, 32);
        l[g][r] = l[g][r] * alpha + rs;
        m[g][r] = mnew[r];
#pragma unroll
        for (int dc = 0; dc < 4; ++dc) Oacc[g][dc][r] *= alpha;
      }
      lds_fence_wave();   // Ps is wave-private; per-wave DS order suffices

      // O += P V   (A = P 16x32, B = V 32x16 per 16-wide d-chunk)
      v16h ap = cat8(*(const v8h*)&Ps[w][lrow][lhi * 8],
                     *(const v8h*)&Ps[w][lrow][16 + lhi * 8]);
#pragma unroll
      for (int dc = 0; dc < 4; ++dc) {
        const int d = dc * 16 + lrow;
        v16h bv = cat8(*(const v8h*)&Vt[d][lhi * 8],
                       *(const v8h*)&Vt[d][16 + lhi * 8]);
        Oacc[g][dc] = wmma16(ap, bv, Oacc[g][dc]);
      }
    }
    __syncthreads();
  }

  // Normalize and write O
#pragma unroll
  for (int g = 0; g < 2; ++g) {
    half_t* orow =
        o + ((size_t)n * SEQ + qb + w * 32 + g * 16) * EMB + h * HDIM;
#pragma unroll
    for (int dc = 0; dc < 4; ++dc)
#pragma unroll
      for (int r = 0; r < 8; ++r) {
        float v = Oacc[g][dc][r] / l[g][r];
        orow[(size_t)(r + lhi * 8) * EMB + dc * 16 + lrow] = (half_t)v;
      }
  }
}

// ---------------------------------------------------------------------------
// Host orchestration
// ---------------------------------------------------------------------------
extern "C" void kernel_launch(void* const* d_in, const int* in_sizes, int n_in,
                              void* d_out, int out_size, void* d_ws,
                              size_t ws_size, hipStream_t stream) {
  const float* x      = (const float*)d_in[0];
  const unsigned char* mask = (const unsigned char*)d_in[1];  // jnp bool = 1B
  const float* ln1_s  = (const float*)d_in[2];
  const float* ln1_b  = (const float*)d_in[3];
  const float* Wqkv   = (const float*)d_in[4];
  const float* bqkv   = (const float*)d_in[5];
  const float* Wo     = (const float*)d_in[6];
  const float* bo     = (const float*)d_in[7];
  const float* ln2_s  = (const float*)d_in[8];
  const float* ln2_b  = (const float*)d_in[9];
  const float* lnf_s  = (const float*)d_in[10];
  const float* lnf_b  = (const float*)d_in[11];
  const float* W1     = (const float*)d_in[12];
  const float* b1     = (const float*)d_in[13];
  const float* W2     = (const float*)d_in[14];
  const float* b2     = (const float*)d_in[15];

  char* ws = (char*)d_ws;
  size_t off = 0;
  auto alloc = [&](size_t bytes) -> void* {
    void* p = ws + off;
    off += (bytes + 255) & ~(size_t)255;
    return p;
  };

  half_t* hf16    = (half_t*)alloc((size_t)ROWS * EMB * 2);
  half_t* WqkvT16 = (half_t*)alloc((size_t)EMB * 3 * EMB * 2);  // [3E][E]
  half_t* WoT16   = (half_t*)alloc((size_t)EMB * EMB * 2);      // [E][E]
  half_t* W1T16   = (half_t*)alloc((size_t)EMB * DFF_ * 2);     // [DFF][E]
  half_t* W2T16   = (half_t*)alloc((size_t)DFF_ * EMB * 2);     // [E][DFF]
  half_t* qkv16   = (half_t*)alloc((size_t)ROWS * 3 * EMB * 2);
  half_t* o16     = (half_t*)alloc((size_t)ROWS * EMB * 2);
  float*  x1      = (float*)alloc((size_t)ROWS * EMB * 4);
  half_t* h216    = (half_t*)alloc((size_t)ROWS * EMB * 2);
  half_t* g16     = (half_t*)alloc((size_t)ROWS * DFF_ * 2);
  (void)in_sizes; (void)n_in; (void)out_size; (void)ws_size;

  auto tcvt = [&](const float* src, half_t* dst, int Kd, int Nd) {
    transpose_f32_to_f16_kernel<<<dim3(Nd / 32, Kd / 32), 256, 0, stream>>>(
        src, dst, Kd, Nd);
  };
  tcvt(Wqkv, WqkvT16, EMB, 3 * EMB);
  tcvt(Wo,   WoT16,   EMB, EMB);
  tcvt(W1,   W1T16,   EMB, DFF_);
  tcvt(W2,   W2T16,   DFF_, EMB);

  // 1) h = LN1(x)
  layernorm_kernel<<<ROWS, 256, 0, stream>>>(x, ln1_s, ln1_b, hf16);

  // 2) qkv = h @ Wqkv + bqkv
  gemm_f16_kernel<EPI_F16><<<dim3(3 * EMB / 128, ROWS / 128), 256, 0, stream>>>(
      hf16, WqkvT16, bqkv, nullptr, qkv16, ROWS, EMB, 3 * EMB);

  // 3) o = attention(q, k, v)
  flash_attn_kernel<<<dim3(SEQ / 128, HEADS, 2), 128, 0, stream>>>(qkv16, mask,
                                                                   o16);

  // 4) x1 = x + o @ Wo + bo
  gemm_f16_kernel<EPI_RES_F32><<<dim3(EMB / 128, ROWS / 128), 256, 0, stream>>>(
      o16, WoT16, bo, x, x1, ROWS, EMB, EMB);

  // 5) h2 = lnf(ln2(x1))
  double_ln_kernel<<<ROWS, 256, 0, stream>>>(x1, ln2_s, ln2_b, lnf_s, lnf_b,
                                             h216);

  // 6) g = gelu(h2 @ W1 + b1)
  gemm_f16_kernel<EPI_GELU_F16><<<dim3(DFF_ / 128, ROWS / 128), 256, 0,
                                  stream>>>(h216, W1T16, b1, nullptr, g16, ROWS,
                                            EMB, DFF_);

  // 7) out = x1 + g @ W2 + b2
  gemm_f16_kernel<EPI_RES_F32><<<dim3(EMB / 128, ROWS / 128), 256, 0, stream>>>(
      g16, W2T16, b2, x1, d_out, ROWS, DFF_, EMB);
}
